// GAT_43739946942551
// MI455X (gfx1250) — matrix operand order repaired and checked by reading, not compiled, not run on hardware
//
#include <hip/hip_runtime.h>
#include <hip/hip_bf16.h>

// ---------------- problem constants ----------------
#define NNODES 50000
#define NEDGES 400000
#define FIN    128
#define NH     8
#define NC     64
#define HC     512            // NH*NC
#define NG     64
#define NCLASS 10
#define ETOT   (NEDGES + NNODES)
#define NEG_SLOPE 0.2f
#define GAT_EPS   1e-16f

typedef __attribute__((ext_vector_type(16))) __bf16          v16bf;
typedef __attribute__((ext_vector_type(16))) unsigned short  v16us;
typedef __attribute__((ext_vector_type(8)))  float           v8f;

__device__ __forceinline__ unsigned short f2bf(float f) {
  // round-to-nearest-even f32 -> bf16
  unsigned u = __float_as_uint(f);
  u += 0x7FFFu + ((u >> 16) & 1u);
  return (unsigned short)(u >> 16);
}

// ---------------- WMMA GEMM: Y[N,HC] = X[N,K] @ W[K,HC] ----------------
// 256 threads = 8 waves arranged 4(M) x 2(N); block tile 64 rows x 64 cols;
// each wave computes two 16x16 WMMA tiles sharing one A fragment.
#define AP 520   // ldsA row stride (ushorts): 512 + 8 pad -> conflict-free b128 reads
#define BP 40    // ldsB row stride (ushorts): 32 + 8 pad, 80B rows (16B aligned)

__global__ __launch_bounds__(256)
void gemm_bf16_wmma(const float* __restrict__ X, const float* __restrict__ W,
                    float* __restrict__ Y, int K) {
  __shared__ unsigned short ldsA[64 * AP];    // 64 x K (bf16), staged once (~66.5 KB @ K=512)
  __shared__ unsigned short ldsB[64 * BP];    // 64 cols x 32 K (bf16, transposed), per k-step
  unsigned* ldsA32 = (unsigned*)ldsA;
  unsigned* ldsB32 = (unsigned*)ldsB;

  const int t     = threadIdx.x;
  const int lane  = t & 31;
  const int wave  = t >> 5;
  const int waveM = wave & 3;
  const int waveN = wave >> 2;
  const int rowBase = blockIdx.x * 64;
  const int colBase = blockIdx.y * 64;

  // stage A tile [64 x K] as packed bf16 pairs (float2 global loads, b32 LDS stores)
  const int Kp = K >> 1;
  for (int p = t; p < 64 * Kp; p += 256) {
    int r = p / Kp, kp = p - r * Kp;
    int row = rowBase + r; if (row >= NNODES) row = NNODES - 1;   // clamp (guarded at store)
    const float* xp = X + (size_t)row * K + 2 * kp;
    unsigned lo = f2bf(xp[0]), hi16 = f2bf(xp[1]);
    ldsA32[r * (AP / 2) + kp] = lo | (hi16 << 16);
  }

  const int m   = lane & 15;     // A: row / B,C,D: column within tile
  const int hi  = lane >> 4;     // lane half
  const int aRow  = (waveM * 16 + m) * AP;
  const int bRow0 = (waveN * 32 + m) * BP;
  const int bRow1 = (waveN * 32 + 16 + m) * BP;
  const int kgrpA = hi * 8;      // A half-lane K offset
  const int kgrpB = hi * 16;     // unused; B fragments use kgrpA (hi*8)
  const int nStage  = t & 63;    // B staging: column
  const int kpStage = t >> 6;    // B staging: k-pair phase (0..3)

  v8f acc0 = {}, acc1 = {};

  for (int kt = 0; kt < K; kt += 32) {
    __syncthreads();  // A ready (iter 0) / previous fragments consumed
    // stage B: W[kt..kt+32) x [colBase..+64) -> ldsB[n][k], packed bf16 pairs
#pragma unroll
    for (int it = 0; it < 4; ++it) {
      int kp = kpStage + it * 4;                 // 0..15
      int k  = 2 * kp;
      const float* wp = W + (size_t)(kt + k) * HC + colBase + nStage;
      unsigned lo = f2bf(wp[0]), hi16 = f2bf(wp[HC]);
      ldsB32[nStage * (BP / 2) + kp] = lo | (hi16 << 16);
    }
    __syncthreads();

    // A fragment: lane m holds K = kt + {kgrpA..+7, 16+kgrpA..+7}
    v16us ua, ub0, ub1;
#pragma unroll
    for (int i = 0; i < 8; ++i) {
      ua[i]     = ldsA[aRow + kt + kgrpA + i];
      ua[i + 8] = ldsA[aRow + kt + 16 + kgrpA + i];
    }
    // B fragments: lane (col m) holds K = {kgrpA..+7, 16+kgrpA..+7}
#pragma unroll
    for (int i = 0; i < 8; ++i) {
      ub0[i]     = ldsB[bRow0 + kgrpA + i];
      ub0[i + 8] = ldsB[bRow0 + 16 + kgrpA + i];
      ub1[i]     = ldsB[bRow1 + kgrpA + i];
      ub1[i + 8] = ldsB[bRow1 + 16 + kgrpA + i];
    }

    v16bf a  = __builtin_bit_cast(v16bf, ua);
    v16bf b0 = __builtin_bit_cast(v16bf, ub0);
    v16bf b1 = __builtin_bit_cast(v16bf, ub1);
    acc0 = __builtin_amdgcn_wmma_f32_16x16x32_bf16(false, a, false, b0,
                                                   (short)0, acc0, false, false);
    acc1 = __builtin_amdgcn_wmma_f32_16x16x32_bf16(false, a, false, b1,
                                                   (short)0, acc1, false, false);
  }

  // D layout: lane half 0 -> rows v, half 1 -> rows v+8; column = m
  const int col = colBase + waveN * 32 + m;
#pragma unroll
  for (int v = 0; v < 8; ++v) {
    int row = rowBase + waveM * 16 + hi * 8 + v;
    if (row < NNODES) {
      Y[(size_t)row * HC + col]      = acc0[v];
      Y[(size_t)row * HC + col + 16] = acc1[v];
    }
  }
}

// ---------------- attention coefficients: a_src/a_dst [N,H] ----------------
__global__ void attn_coeff(const float* __restrict__ h,
                           const float* __restrict__ as,
                           const float* __restrict__ ad,
                           float* __restrict__ os, float* __restrict__ od) {
  int gid = blockIdx.x * blockDim.x + threadIdx.x;
  if (gid >= NNODES * NH) return;
  int n = gid >> 3, hd = gid & 7;
  const float* hp = h + (size_t)n * HC + hd * NC;
  const float* sp = as + hd * NC;
  const float* dp = ad + hd * NC;
  float ss = 0.f, sd = 0.f;
#pragma unroll
  for (int c = 0; c < NC; ++c) { float v = hp[c]; ss += v * sp[c]; sd += v * dp[c]; }
  os[gid] = ss; od[gid] = sd;
}

// ---------------- generic fill ----------------
__global__ void fill_u32(unsigned* __restrict__ p, unsigned val, int n) {
  int i = blockIdx.x * blockDim.x + threadIdx.x;
  if (i < n) p[i] = val;
}

// ---------------- ordered-uint float max helpers ----------------
__device__ __forceinline__ unsigned ford(float f) {
  unsigned u = __float_as_uint(f);
  return (u & 0x80000000u) ? ~u : (u | 0x80000000u);
}
__device__ __forceinline__ float funord(unsigned u) {
  return (u & 0x80000000u) ? __uint_as_float(u ^ 0x80000000u) : __uint_as_float(~u);
}

// ---------------- edge pass 1: leaky-relu + segment max ----------------
__global__ void edge_pass1(const int* __restrict__ ei,
                           const float* __restrict__ as, const float* __restrict__ ad,
                           float* __restrict__ alpha, unsigned* __restrict__ amax) {
  int gid = blockIdx.x * blockDim.x + threadIdx.x;
  if (gid >= ETOT * NH) return;
  int e = gid >> 3, hd = gid & 7;
  int s, d;
  if (e < NEDGES) { s = ei[e]; d = ei[NEDGES + e]; } else { s = d = e - NEDGES; }
  float v = as[s * NH + hd] + ad[d * NH + hd];
  v = (v >= 0.f) ? v : NEG_SLOPE * v;
  alpha[gid] = v;
  atomicMax(&amax[d * NH + hd], ford(v));
}

// ---------------- edge pass 2: exp + segment sum ----------------
__global__ void edge_pass2(const int* __restrict__ ei, float* __restrict__ alpha,
                           const unsigned* __restrict__ amax, float* __restrict__ denom) {
  int gid = blockIdx.x * blockDim.x + threadIdx.x;
  if (gid >= ETOT * NH) return;
  int e = gid >> 3, hd = gid & 7;
  int d = (e < NEDGES) ? ei[NEDGES + e] : e - NEDGES;
  float ex = __expf(alpha[gid] - funord(amax[d * NH + hd]));
  alpha[gid] = ex;
  atomicAdd(&denom[d * NH + hd], ex);
}

// ---------------- edge pass 3: weighted message scatter-add ----------------
__global__ __launch_bounds__(256)
void edge_agg(const int* __restrict__ ei, const float* __restrict__ h,
              const float* __restrict__ alpha, const float* __restrict__ denom,
              float* __restrict__ out) {
  int e = blockIdx.x;
  int s, d;
  if (e < NEDGES) { s = ei[e]; d = ei[NEDGES + e]; } else { s = d = e - NEDGES; }
#pragma unroll
  for (int half = 0; half < 2; ++half) {
    int chan = threadIdx.x + half * 256;
    int hd = chan >> 6;
    float w = alpha[(size_t)e * NH + hd] / (denom[d * NH + hd] + GAT_EPS);
    atomicAdd(&out[(size_t)d * HC + chan], w * h[(size_t)s * HC + chan]);
  }
}

// ---------------- bias + ReLU (concat layers) ----------------
__global__ void bias_relu(float* __restrict__ y, const float* __restrict__ b, int n) {
  int i = blockIdx.x * blockDim.x + threadIdx.x;
  if (i >= n) return;
  float v = y[i] + b[i & (HC - 1)];
  y[i] = v > 0.f ? v : 0.f;
}

// ---------------- head mean + bias + ReLU (layer 3) ----------------
__global__ void head_mean_bias_relu(const float* __restrict__ agg,
                                    const float* __restrict__ b, float* __restrict__ h3) {
  int gid = blockIdx.x * blockDim.x + threadIdx.x;
  if (gid >= NNODES * NC) return;
  int n = gid >> 6, c = gid & 63;
  float s = 0.f;
#pragma unroll
  for (int hd = 0; hd < NH; ++hd) s += agg[(size_t)n * HC + hd * NC + c];
  float v = s * (1.0f / NH) + b[c];
  h3[gid] = v > 0.f ? v : 0.f;
}

// ---------------- global mean pool (segment sum) ----------------
__global__ void pool_sum(const float* __restrict__ h3, const int* __restrict__ batch,
                         float* __restrict__ sums, float* __restrict__ cnts) {
  int gid = blockIdx.x * blockDim.x + threadIdx.x;
  if (gid >= NNODES * NC) return;
  int n = gid >> 6, c = gid & 63;
  int g = batch[n];
  atomicAdd(&sums[g * NC + c], h3[gid]);
  if (c == 0) atomicAdd(&cnts[g], 1.0f);
}

// ---------------- FC + log_softmax ----------------
__global__ void fc_logsoftmax(const float* __restrict__ sums, const float* __restrict__ cnts,
                              const float* __restrict__ fcw, const float* __restrict__ fcb,
                              float* __restrict__ out) {
  int g = threadIdx.x;
  if (g >= NG) return;
  float inv = 1.0f / fmaxf(cnts[g], 1.0f);
  float logits[NCLASS];
#pragma unroll
  for (int j = 0; j < NCLASS; ++j) logits[j] = fcb[j];
  for (int c = 0; c < NC; ++c) {
    float p = sums[g * NC + c] * inv;
#pragma unroll
    for (int j = 0; j < NCLASS; ++j) logits[j] += p * fcw[c * NCLASS + j];
  }
  float mx = logits[0];
#pragma unroll
  for (int j = 1; j < NCLASS; ++j) mx = fmaxf(mx, logits[j]);
  float se = 0.f;
#pragma unroll
  for (int j = 0; j < NCLASS; ++j) se += __expf(logits[j] - mx);
  float lse = mx + __logf(se);
#pragma unroll
  for (int j = 0; j < NCLASS; ++j) out[g * NCLASS + j] = logits[j] - lse;
}

// ---------------- host launcher ----------------
extern "C" void kernel_launch(void* const* d_in, const int* in_sizes, int n_in,
                              void* d_out, int out_size, void* d_ws, size_t ws_size,
                              hipStream_t stream) {
  const float* x    = (const float*)d_in[0];
  const int*   ei   = (const int*)d_in[1];
  const int*   bat  = (const int*)d_in[2];
  const float* W1   = (const float*)d_in[3];
  const float* a1s  = (const float*)d_in[4];
  const float* a1d  = (const float*)d_in[5];
  const float* b1   = (const float*)d_in[6];
  const float* W2   = (const float*)d_in[7];
  const float* a2s  = (const float*)d_in[8];
  const float* a2d  = (const float*)d_in[9];
  const float* b2   = (const float*)d_in[10];
  const float* W3   = (const float*)d_in[11];
  const float* a3s  = (const float*)d_in[12];
  const float* a3d  = (const float*)d_in[13];
  const float* b3   = (const float*)d_in[14];
  const float* fcw  = (const float*)d_in[15];
  const float* fcb  = (const float*)d_in[16];
  float* out = (float*)d_out;

  // workspace layout (~226 MB)
  float*    ws    = (float*)d_ws;
  float*    hbuf0 = ws;                                      // N*HC
  float*    hbuf1 = hbuf0 + (size_t)NNODES * HC;             // N*HC
  float*    alpha = hbuf1 + (size_t)NNODES * HC;             // ETOT*NH
  float*    asrc  = alpha + (size_t)ETOT * NH;               // N*NH
  float*    adst  = asrc + (size_t)NNODES * NH;              // N*NH
  unsigned* amax  = (unsigned*)(adst + (size_t)NNODES * NH); // N*NH
  float*    denom = (float*)(amax + (size_t)NNODES * NH);    // N*NH
  float*    sums  = denom + (size_t)NNODES * NH;             // NG*NC
  float*    cnts  = sums + NG * NC;                          // NG

  const int NHn  = NNODES * NH;
  const int EHn  = ETOT * NH;
  const int NHCn = NNODES * HC;
  const int NNCn = NNODES * NC;
  dim3 ggrid((NNODES + 63) / 64, HC / 64);

  auto layer = [&](const float* Xin, int K, const float* Wm,
                   const float* asW, const float* adW) {
    gemm_bf16_wmma<<<ggrid, 256, 0, stream>>>(Xin, Wm, hbuf1, K);
    attn_coeff<<<(NHn + 255) / 256, 256, 0, stream>>>(hbuf1, asW, adW, asrc, adst);
    fill_u32<<<(NHn + 255) / 256, 256, 0, stream>>>(amax, 0u, NHn);
    fill_u32<<<(NHn + 255) / 256, 256, 0, stream>>>((unsigned*)denom, 0u, NHn);
    edge_pass1<<<(EHn + 255) / 256, 256, 0, stream>>>(ei, asrc, adst, alpha, amax);
    edge_pass2<<<(EHn + 255) / 256, 256, 0, stream>>>(ei, alpha, amax, denom);
    fill_u32<<<(NHCn + 255) / 256, 256, 0, stream>>>((unsigned*)hbuf0, 0u, NHCn);
    edge_agg<<<ETOT, 256, 0, stream>>>(ei, hbuf1, alpha, denom, hbuf0);
  };

  // layer 1: x[N,128] -> hbuf0[N,512]
  layer(x, FIN, W1, a1s, a1d);
  bias_relu<<<(NHCn + 255) / 256, 256, 0, stream>>>(hbuf0, b1, NHCn);
  // layer 2
  layer(hbuf0, HC, W2, a2s, a2d);
  bias_relu<<<(NHCn + 255) / 256, 256, 0, stream>>>(hbuf0, b2, NHCn);
  // layer 3 (mean over heads)
  layer(hbuf0, HC, W3, a3s, a3d);
  head_mean_bias_relu<<<(NNCn + 255) / 256, 256, 0, stream>>>(hbuf0, b3, hbuf1);

  // pool + classify
  fill_u32<<<(NG * NC + NG + 255) / 256, 256, 0, stream>>>((unsigned*)sums, 0u, NG * NC + NG);
  pool_sum<<<(NNCn + 255) / 256, 256, 0, stream>>>(hbuf1, bat, sums, cnts);
  fc_logsoftmax<<<1, NG, 0, stream>>>(sums, cnts, fcw, fcb, out);

  (void)in_sizes; (void)n_in; (void)out_size; (void)ws_size;
}
